// TemporalGNN_84293028152001
// MI455X (gfx1250) — compile-verified, matrix operand order checked by reading, not run on hardware
//
#include <hip/hip_runtime.h>
#include <hip/hip_bf16.h>

// ---------------------------------------------------------------------------
// TemporalGNN for MI455X (gfx1250):
//   - all GEMMs via v_wmma_f32_16x16x32_bf16 (fp32 accumulate)
//   - B (weight) tiles staged block-wide into LDS via TDM tensor_load_to_lds
//   - A operand software-pipelined from global memory
// ---------------------------------------------------------------------------

#define N_NODES 10000
#define N_EDGES 320000
#define F_DIM   64
#define H_DIM   256
#define G_DIM   256

#define USE_TDM 1   // 1: TDM tensor_load_to_lds staging; 0: plain cooperative load

typedef __attribute__((ext_vector_type(16))) __bf16 v16bf;
typedef __attribute__((ext_vector_type(8)))  float  v8f;
typedef unsigned int v4u __attribute__((ext_vector_type(4)));
typedef unsigned int v8u __attribute__((ext_vector_type(8)));

union FragBF { v16bf v; uint4 q[2]; };

__device__ __forceinline__ unsigned short f2bf(float f) {
    union { float f; unsigned int u; } x; x.f = f;
    unsigned int u = x.u;
    unsigned int r = u + 0x7FFFu + ((u >> 16) & 1u);   // round-to-nearest-even
    return (unsigned short)(r >> 16);
}

__device__ __forceinline__ float sigmoidf_(float x) { return 1.f / (1.f + expf(-x)); }

__device__ __forceinline__ void atomicMaxF(float* addr, float val) {
    int* ia = (int*)addr;
    int old = __float_as_int(*addr);
    while (__int_as_float(old) < val) {
        int assumed = old;
        old = atomicCAS(ia, assumed, __float_as_int(val));
        if (old == assumed) break;
    }
}

// ---------------- utility kernels ----------------

__global__ void k_fill_f32(float* p, float v, long n) {
    long i = (long)blockIdx.x * blockDim.x + threadIdx.x;
    long st = (long)gridDim.x * blockDim.x;
    for (; i < n; i += st) p[i] = v;
}

__global__ void k_fill_u16(unsigned short* p, unsigned short v, long n) {
    long i = (long)blockIdx.x * blockDim.x + threadIdx.x;
    long st = (long)gridDim.x * blockDim.x;
    for (; i < n; i += st) p[i] = v;
}

__global__ void k_f2bf_arr(const float* __restrict__ in, unsigned short* __restrict__ out, long n) {
    long i = (long)blockIdx.x * blockDim.x + threadIdx.x;
    long st = (long)gridDim.x * blockDim.x;
    for (; i < n; i += st) out[i] = f2bf(in[i]);
}

// ---------------- WMMA GEMM:  C[M,N] = act( A[M,K]bf16 @ B[N,K]bf16^T + bias ) ----------------
// Block = 256 threads = 8 waves. All waves share ONE 64-column B tile (staged
// in LDS by the TDM); each wave owns a 16-row M strip -> block tile 128x64.
// A-frag (16x32 bf16): lanes 0-15 -> row m0+lane, K {k..k+7, k+16..k+23}
//                      lanes 16-31 -> row m0+lane-16, K {k+8..k+15, k+24..k+31}
// B-frag mirrors A with output column in place of row (B row n = weight row n).
__global__ __launch_bounds__(256) void k_gemm_bf16(
    const unsigned short* __restrict__ A, int lda,
    const unsigned short* __restrict__ B, int ldb,
    float* __restrict__ C, int ldc,
    const float* __restrict__ bias,
    unsigned short* __restrict__ Cbf, int ldcbf, int cbfoff,
    int M, int N, int K, int act)
{
    extern __shared__ unsigned short lb[];          // [64][K] bf16 B tile

    const int tilesN  = N >> 6;
    const int blocksM = ((M >> 4) + 7) >> 3;
    const int bN = blockIdx.x / blocksM;
    const int bM = blockIdx.x - bN * blocksM;
    const int n0 = bN * 64;
    (void)tilesN;

    // ---- stage B tile (rows n0..n0+63, K columns) into LDS ----
#if USE_TDM
    if ((threadIdx.x >> 5) == 0) {
        unsigned long long ga = (unsigned long long)(const void*)(B + (size_t)n0 * ldb);
        unsigned lds_off = (unsigned)(unsigned long long)
            (__attribute__((address_space(3))) unsigned short*)lb;
        // D# group0 (4 dwords): count=1 | lds_addr | global_addr | type=2
        v4u g0;
        g0[0] = 1u;
        g0[1] = lds_off;
        g0[2] = (unsigned)ga;
        g0[3] = (unsigned)((ga >> 32) & 0x01FFFFFFu) | (2u << 30);
        // D# group1 (8 dwords): data_size=1(2B); tensor_dim0=K; tensor_dim1=64;
        // tile_dim0=K; tile_dim1=64; tensor_dim0_stride=ldb
        v8u g1;
        g1[0] = (1u << 16);
        g1[1] = ((unsigned)K & 0xFFFFu) << 16;                    // tensor_dim0 lo
        g1[2] = (((unsigned)K >> 16) & 0xFFFFu) | (64u << 16);    // dim0 hi | tensor_dim1 lo
        g1[3] = ((unsigned)K & 0xFFFFu) << 16;                    // dim1 hi=0 | tile_dim0=K
        g1[4] = 64u;                                              // tile_dim1=64, tile_dim2=0
        g1[5] = (unsigned)ldb;                                    // stride0 lo
        g1[6] = 0u;                                               // stride0 hi | stride1 lo
        g1[7] = 0u;                                               // stride1 hi
        asm volatile("tensor_load_to_lds %0, %1" :: "s"(g0), "s"(g1) : "memory");
        __builtin_amdgcn_s_wait_tensorcnt(0);
    }
#else
    {
        const int chunks = (64 * K) >> 3;                 // 8-ushort chunks
        const int kc8 = K >> 3;
        for (int c = threadIdx.x; c < chunks; c += 256) {
            int r = c / kc8, kc = (c - r * kc8) << 3;
            *(uint4*)&lb[r * K + kc] = *(const uint4*)(B + (size_t)(n0 + r) * ldb + kc);
        }
    }
#endif
    __syncthreads();

    const int wave = threadIdx.x >> 5;
    const int lane = threadIdx.x & 31;
    const int m0 = (bM * 8 + wave) * 16;
    if (m0 >= M) return;

    const int lr = lane & 15;
    const int hi = lane >> 4;       // K-subgroup select

    const unsigned short* arow = A + (size_t)(m0 + lr) * lda + hi * 8;
    const unsigned short* b0 = &lb[(size_t)(lr)      * K + hi * 8];
    const unsigned short* b1 = &lb[(size_t)(16 + lr) * K + hi * 8];
    const unsigned short* b2 = &lb[(size_t)(32 + lr) * K + hi * 8];
    const unsigned short* b3 = &lb[(size_t)(48 + lr) * K + hi * 8];

    v8f acc[4] = {};
    FragBF a;
    a.q[0] = *(const uint4*)(arow);
    a.q[1] = *(const uint4*)(arow + 16);

    int k = 0;
    for (; k + 32 < K; k += 32) {
        FragBF an;                                    // prefetch next A (global)
        an.q[0] = *(const uint4*)(arow + k + 32);
        an.q[1] = *(const uint4*)(arow + k + 48);
        FragBF b[4];                                  // current B (LDS)
        b[0].q[0] = *(const uint4*)(b0 + k); b[0].q[1] = *(const uint4*)(b0 + k + 16);
        b[1].q[0] = *(const uint4*)(b1 + k); b[1].q[1] = *(const uint4*)(b1 + k + 16);
        b[2].q[0] = *(const uint4*)(b2 + k); b[2].q[1] = *(const uint4*)(b2 + k + 16);
        b[3].q[0] = *(const uint4*)(b3 + k); b[3].q[1] = *(const uint4*)(b3 + k + 16);
#pragma unroll
        for (int j = 0; j < 4; ++j)
            acc[j] = __builtin_amdgcn_wmma_f32_16x16x32_bf16(
                false, a.v, false, b[j].v, (short)0, acc[j], false, false);
        a = an;
    }
    {
        FragBF b[4];
        b[0].q[0] = *(const uint4*)(b0 + k); b[0].q[1] = *(const uint4*)(b0 + k + 16);
        b[1].q[0] = *(const uint4*)(b1 + k); b[1].q[1] = *(const uint4*)(b1 + k + 16);
        b[2].q[0] = *(const uint4*)(b2 + k); b[2].q[1] = *(const uint4*)(b2 + k + 16);
        b[3].q[0] = *(const uint4*)(b3 + k); b[3].q[1] = *(const uint4*)(b3 + k + 16);
#pragma unroll
        for (int j = 0; j < 4; ++j)
            acc[j] = __builtin_amdgcn_wmma_f32_16x16x32_bf16(
                false, a.v, false, b[j].v, (short)0, acc[j], false, false);
    }

    const int rbase = m0 + hi * 8;
#pragma unroll
    for (int j = 0; j < 4; ++j) {
        const int col = n0 + 16 * j + lr;
        const float bv = bias ? bias[col] : 0.f;
#pragma unroll
        for (int r = 0; r < 8; ++r) {
            float v = acc[j][r] + bv;
            if (act == 1) v = fmaxf(v, 0.f);
            const int row = rbase + r;
            C[(size_t)row * ldc + col] = v;
            if (Cbf) Cbf[(size_t)row * ldcbf + cbfoff + col] = f2bf(v);
        }
    }
}

// ---------------- GRU gate fusion (one element of H per thread) ----------------
__global__ void k_gru_gate(const float* __restrict__ ih, const float* __restrict__ hh,
                           float* __restrict__ h, unsigned short* __restrict__ hcomb, int hoff)
{
    int idx = blockIdx.x * blockDim.x + threadIdx.x;
    if (idx >= N_NODES * H_DIM) return;
    int node = idx >> 8, j = idx & 255;
    const float* ip = ih + (size_t)node * 768;
    const float* hp = hh + (size_t)node * 768;
    float r = sigmoidf_(ip[j] + hp[j]);
    float z = sigmoidf_(ip[256 + j] + hp[256 + j]);
    float nv = tanhf(ip[512 + j] + r * hp[512 + j]);
    float hprev = h[(size_t)node * 256 + j];
    float hnew = (1.f - z) * nv + z * hprev;
    h[(size_t)node * 256 + j] = hnew;
    hcomb[(size_t)node * 768 + hoff + j] = f2bf(hnew);
}

// ---------------- SAGE attention helpers ----------------
__global__ void k_node_attn(const float* __restrict__ xl, const float* __restrict__ aw,
                            float* __restrict__ p1, float* __restrict__ p2)
{
    int n = blockIdx.x * blockDim.x + threadIdx.x;
    if (n >= N_NODES) return;
    const float* xr = xl + (size_t)n * G_DIM;
    float s1 = 0.f, s2 = 0.f;
#pragma unroll 4
    for (int g = 0; g < G_DIM; ++g) { float v = xr[g]; s1 += v * aw[g]; s2 += v * aw[G_DIM + g]; }
    p1[n] = s1; p2[n] = s2;
}

__global__ void k_edge_score(const int* __restrict__ row, const int* __restrict__ col,
                             const float* __restrict__ ew, const float* __restrict__ p1,
                             const float* __restrict__ p2, const float* __restrict__ aw,
                             const float* __restrict__ ab,
                             float* __restrict__ segmax, float* __restrict__ esc)
{
    int e = blockIdx.x * blockDim.x + threadIdx.x;
    if (e >= N_EDGES) return;
    int r = row[e], c = col[e];
    float s = p1[r] + p2[c] + ew[e] * aw[2 * G_DIM] + ab[0];
    esc[e] = s;
    atomicMaxF(&segmax[r], s);
}

__global__ void k_edge_exp(const int* __restrict__ row, float* __restrict__ esc,
                           const float* __restrict__ segmax, float* __restrict__ segsum)
{
    int e = blockIdx.x * blockDim.x + threadIdx.x;
    if (e >= N_EDGES) return;
    int r = row[e];
    float v = expf(esc[e] - segmax[r]);
    esc[e] = v;
    atomicAdd(&segsum[r], v);
}

__global__ void k_edge_alpha(const int* __restrict__ row, float* __restrict__ esc,
                             const float* __restrict__ segsum)
{
    int e = blockIdx.x * blockDim.x + threadIdx.x;
    if (e >= N_EDGES) return;
    esc[e] = esc[e] / (segsum[row[e]] + 1e-16f);
}

__global__ void k_edge_scatter(const int* __restrict__ row, const int* __restrict__ col,
                               const float* __restrict__ alpha, const float* __restrict__ xl,
                               float* __restrict__ agg)
{
    long i = (long)blockIdx.x * blockDim.x + threadIdx.x;   // N_EDGES * 256 threads
    int e = (int)(i >> 8), g = (int)(i & 255);
    int r = row[e], c = col[e];
    atomicAdd(&agg[(size_t)c * G_DIM + g], alpha[e] * xl[(size_t)r * G_DIM + g]);
}

__global__ void k_post1(const float* __restrict__ agg, float* __restrict__ g1,
                        unsigned short* __restrict__ g1bf)
{
    long i = (long)blockIdx.x * blockDim.x + threadIdx.x;
    if (i >= (long)N_NODES * G_DIM) return;
    float v = fmaxf(agg[i], 0.f);
    g1[i] = v;
    g1bf[i] = f2bf(v);
}

__global__ void k_post2(const float* __restrict__ agg, const float* __restrict__ temporal,
                        float* __restrict__ g2)
{
    long i = (long)blockIdx.x * blockDim.x + threadIdx.x;
    if (i >= (long)N_NODES * G_DIM) return;
    g2[i] = fmaxf(agg[i], 0.f) + temporal[i];
}

__global__ void k_heads(const float* __restrict__ g2,
                        const float* __restrict__ impW, const float* __restrict__ impb,
                        const float* __restrict__ uncW, const float* __restrict__ uncb,
                        float* __restrict__ out)
{
    int n = blockIdx.x * blockDim.x + threadIdx.x;
    if (n >= N_NODES) return;
    const float* gr = g2 + (size_t)n * G_DIM;
    float si = impb[0], su = uncb[0];
#pragma unroll 4
    for (int g = 0; g < G_DIM; ++g) { float v = gr[g]; si += v * impW[g]; su += v * uncW[g]; }
    out[n] = tanhf(si);
    out[N_NODES + n] = sigmoidf_(su);
}

// ---------------------------------------------------------------------------

extern "C" void kernel_launch(void* const* d_in, const int* in_sizes, int n_in,
                              void* d_out, int out_size, void* d_ws, size_t ws_size,
                              hipStream_t stream)
{
    (void)in_sizes; (void)n_in; (void)out_size; (void)ws_size;

    // ---- inputs (setup_inputs order) ----
    const float* xs32  = (const float*)d_in[0];   // [N, 8, 64]
    const float* xm32  = (const float*)d_in[1];   // [N, 30, 64]
    const float* xl32f = (const float*)d_in[2];   // [N, 60, 64]
    const int*   ei    = (const int*)d_in[3];     // [2, E]
    const float* ew    = (const float*)d_in[4];   // [E]
    const float* sWih = (const float*)d_in[5],  *sWhh = (const float*)d_in[6];
    const float* sbih = (const float*)d_in[7],  *sbhh = (const float*)d_in[8];
    const float* mWih = (const float*)d_in[9],  *mWhh = (const float*)d_in[10];
    const float* mbih = (const float*)d_in[11], *mbhh = (const float*)d_in[12];
    const float* lWih = (const float*)d_in[13], *lWhh = (const float*)d_in[14];
    const float* lbih = (const float*)d_in[15], *lbhh = (const float*)d_in[16];
    const float* fW   = (const float*)d_in[17], *fb   = (const float*)d_in[18];
    const float* g1W  = (const float*)d_in[19], *g1aw = (const float*)d_in[20], *g1ab = (const float*)d_in[21];
    const float* g2W  = (const float*)d_in[22], *g2aw = (const float*)d_in[23], *g2ab = (const float*)d_in[24];
    const float* impW = (const float*)d_in[25], *impb = (const float*)d_in[26];
    const float* uncW = (const float*)d_in[27], *uncb = (const float*)d_in[28];
    float* out = (float*)d_out;
    const int* rowp = ei;
    const int* colp = ei + N_EDGES;

    // ---- workspace bump allocator (256B aligned) ----
    size_t off = 0;
    auto alloc = [&](size_t bytes) -> void* {
        void* p = (char*)d_ws + off;
        off += (bytes + 255) & ~(size_t)255;
        return p;
    };
    unsigned short* xs16   = (unsigned short*)alloc((size_t)N_NODES * 8  * F_DIM * 2);
    unsigned short* xm16   = (unsigned short*)alloc((size_t)N_NODES * 30 * F_DIM * 2);
    unsigned short* xl16   = (unsigned short*)alloc((size_t)N_NODES * 60 * F_DIM * 2);
    unsigned short* sWih16 = (unsigned short*)alloc(768 * 64 * 2);
    unsigned short* sWhh16 = (unsigned short*)alloc(768 * 256 * 2);
    unsigned short* mWih16 = (unsigned short*)alloc(768 * 64 * 2);
    unsigned short* mWhh16 = (unsigned short*)alloc(768 * 256 * 2);
    unsigned short* lWih16 = (unsigned short*)alloc(768 * 64 * 2);
    unsigned short* lWhh16 = (unsigned short*)alloc(768 * 256 * 2);
    unsigned short* fW16   = (unsigned short*)alloc(256 * 768 * 2);
    unsigned short* g1W16  = (unsigned short*)alloc(256 * 256 * 2);
    unsigned short* g2W16  = (unsigned short*)alloc(256 * 256 * 2);
    float*          h32    = (float*)alloc((size_t)N_NODES * H_DIM * 4);
    unsigned short* hcomb  = (unsigned short*)alloc((size_t)N_NODES * 768 * 2);
    float*          ihbuf  = (float*)alloc((size_t)N_NODES * 768 * 4);
    float*          hhbuf  = (float*)alloc((size_t)N_NODES * 768 * 4);
    float*          temp32 = (float*)alloc((size_t)N_NODES * G_DIM * 4);
    unsigned short* temp16 = (unsigned short*)alloc((size_t)N_NODES * G_DIM * 2);
    float*          lin32  = (float*)alloc((size_t)N_NODES * G_DIM * 4);
    float*          p1     = (float*)alloc(N_NODES * 4);
    float*          p2     = (float*)alloc(N_NODES * 4);
    float*          segmax = (float*)alloc(N_NODES * 4);
    float*          segsum = (float*)alloc(N_NODES * 4);
    float*          esc    = (float*)alloc(N_EDGES * 4);
    float*          agg    = (float*)alloc((size_t)N_NODES * G_DIM * 4);
    float*          g1_32  = (float*)alloc((size_t)N_NODES * G_DIM * 4);
    unsigned short* g1_16  = (unsigned short*)alloc((size_t)N_NODES * G_DIM * 2);
    float*          g2_32  = (float*)alloc((size_t)N_NODES * G_DIM * 4);

    auto cvt = [&](const float* src, unsigned short* dst, long n) {
        k_f2bf_arr<<<1024, 256, 0, stream>>>(src, dst, n);
    };
    auto gemm = [&](const unsigned short* A, int lda, const unsigned short* B, int ldb,
                    float* C, int ldc, const float* bias,
                    unsigned short* Cbf, int ldcbf, int cbfoff,
                    int M, int N, int K, int act) {
        int blocksM = ((M / 16) + 7) / 8;
        int blocks  = (N / 64) * blocksM;
        size_t shmem = (size_t)64 * K * 2;                 // B tile in LDS
        k_gemm_bf16<<<blocks, 256, shmem, stream>>>(A, lda, B, ldb, C, ldc, bias,
                                                    Cbf, ldcbf, cbfoff, M, N, K, act);
    };

    // ---- precision conversions (one pass each) ----
    cvt(xs32,  xs16, (long)N_NODES * 8  * F_DIM);
    cvt(xm32,  xm16, (long)N_NODES * 30 * F_DIM);
    cvt(xl32f, xl16, (long)N_NODES * 60 * F_DIM);
    cvt(sWih, sWih16, 768 * 64);  cvt(sWhh, sWhh16, 768 * 256);
    cvt(mWih, mWih16, 768 * 64);  cvt(mWhh, mWhh16, 768 * 256);
    cvt(lWih, lWih16, 768 * 64);  cvt(lWhh, lWhh16, 768 * 256);
    cvt(fW, fW16, 256 * 768); cvt(g1W, g1W16, 256 * 256); cvt(g2W, g2W16, 256 * 256);

    // h-combined bf16 buffer starts at zero (h0 == 0)
    k_fill_u16<<<1024, 256, 0, stream>>>(hcomb, 0, (long)N_NODES * 768);

    // ---- three GRUs, writing final hidden into column blocks of hcomb ----
    auto run_gru = [&](const unsigned short* xbf, int T,
                       const unsigned short* Wih16, const unsigned short* Whh16,
                       const float* bih, const float* bhh, int goff) {
        k_fill_f32<<<1024, 256, 0, stream>>>(h32, 0.f, (long)N_NODES * H_DIM);
        for (int t = 0; t < T; ++t) {
            // ih = x_t @ W_ih^T + b_ih   (M=N_NODES, N=768, K=64)
            gemm(xbf + t * F_DIM, T * F_DIM, Wih16, F_DIM,
                 ihbuf, 768, bih, nullptr, 0, 0, N_NODES, 768, F_DIM, 0);
            // hh = h @ W_hh^T + b_hh     (M=N_NODES, N=768, K=256)
            gemm(hcomb + goff, 768, Whh16, H_DIM,
                 hhbuf, 768, bhh, nullptr, 0, 0, N_NODES, 768, H_DIM, 0);
            k_gru_gate<<<(N_NODES * H_DIM + 255) / 256, 256, 0, stream>>>(
                ihbuf, hhbuf, h32, hcomb, goff);
        }
    };
    run_gru(xs16, 8,  sWih16, sWhh16, sbih, sbhh, 0);
    run_gru(xm16, 30, mWih16, mWhh16, mbih, mbhh, 256);
    run_gru(xl16, 60, lWih16, lWhh16, lbih, lbhh, 512);

    // ---- fusion: temporal = relu(concat(hs,hm,hl) @ fW^T + fb) ----
    gemm(hcomb, 768, fW16, 768, temp32, G_DIM, fb, temp16, G_DIM, 0,
         N_NODES, G_DIM, 768, 1);

    // ---- SAGE attention layer ----
    auto run_sage = [&](const unsigned short* xin16, const unsigned short* W16,
                        const float* aw, const float* ab) {
        gemm(xin16, G_DIM, W16, G_DIM, lin32, G_DIM, nullptr, nullptr, 0, 0,
             N_NODES, G_DIM, G_DIM, 0);
        k_node_attn<<<(N_NODES + 255) / 256, 256, 0, stream>>>(lin32, aw, p1, p2);
        k_fill_f32<<<1024, 256, 0, stream>>>(segmax, -3e38f, N_NODES);
        k_fill_f32<<<1024, 256, 0, stream>>>(segsum, 0.f, N_NODES);
        k_fill_f32<<<1024, 256, 0, stream>>>(agg, 0.f, (long)N_NODES * G_DIM);
        k_edge_score<<<(N_EDGES + 255) / 256, 256, 0, stream>>>(
            rowp, colp, ew, p1, p2, aw, ab, segmax, esc);
        k_edge_exp<<<(N_EDGES + 255) / 256, 256, 0, stream>>>(rowp, esc, segmax, segsum);
        k_edge_alpha<<<(N_EDGES + 255) / 256, 256, 0, stream>>>(rowp, esc, segsum);
        k_edge_scatter<<<N_EDGES, 256, 0, stream>>>(rowp, colp, esc, lin32, agg);
    };

    run_sage(temp16, g1W16, g1aw, g1ab);
    k_post1<<<(N_NODES * G_DIM + 255) / 256, 256, 0, stream>>>(agg, g1_32, g1_16);

    run_sage(g1_16, g2W16, g2aw, g2ab);
    k_post2<<<(N_NODES * G_DIM + 255) / 256, 256, 0, stream>>>(agg, temp32, g2_32);

    // ---- heads ----
    k_heads<<<(N_NODES + 255) / 256, 256, 0, stream>>>(g2_32, impW, impb, uncW, uncb, out);
}
